// RelationModel_double_38259568673081
// MI455X (gfx1250) — matrix operand order repaired and checked by reading, not compiled, not run on hardware
//
#include <hip/hip_runtime.h>

// ---------------------------------------------------------------------------
// CDNA5 (gfx1250) fused dual-stage 4-token-group attention ("RelationModel").
// bf16 WMMA 16x16x32 path; 4 positions packed into M=16 rows per wave.
// Round 2: native bf16 converts (v_cvt) + 2-tile WMMA batching for ILP.
// ---------------------------------------------------------------------------

typedef __attribute__((ext_vector_type(16))) __bf16 v16bf;
typedef __attribute__((ext_vector_type(2)))  __bf16 v2bf;
typedef __attribute__((ext_vector_type(8)))  float  v8f;

#define NB      8
#define NP      3136
#define DIMD    256
#define QKVN    768
#define NPOS    (NB * NP)          // 25088
#define OUTPLANE ((size_t)NPOS * DIMD)
#define SCALE_F 0.17677669529663687f   // 32^-0.5

// Native HW converts (RNE) instead of manual bit twiddling.
__device__ __forceinline__ unsigned short bf1(float f) {
  __bf16 b = (__bf16)f;
  return __builtin_bit_cast(unsigned short, b);
}
__device__ __forceinline__ unsigned pack_bf16(float lo, float hi) {
  v2bf v = { (__bf16)lo, (__bf16)hi };          // -> v_cvt_pk_bf16_f32
  return __builtin_bit_cast(unsigned, v);
}
__device__ __forceinline__ float bflo(unsigned u) { return __uint_as_float(u << 16); }
__device__ __forceinline__ float bfhi(unsigned u) { return __uint_as_float(u & 0xFFFF0000u); }

union AF { v16bf v; unsigned u[8]; };

// ---------------------------------------------------------------------------
// Weight pre-pack: W (N x 256 row-major f32) -> bf16 B-fragment layout.
// dst[((t*8+kc)*32 + lane)*16 + e] = bf16( W[n*256 + k] )
//   n = t*16 + (lane&15),  k = kc*32 + (e>>3)*16 + (lane>>4)*8 + (e&7)
// ---------------------------------------------------------------------------
__global__ void pack_weights(const float* __restrict__ W,
                             unsigned short* __restrict__ dst, int ntiles) {
  int o = blockIdx.x * blockDim.x + threadIdx.x;
  int total = ntiles * 8 * 32 * 16;
  if (o >= total) return;
  int e    = o & 15;
  int lane = (o >> 4) & 31;
  int kc   = (o >> 9) & 7;
  int t    = o >> 12;
  int n = t * 16 + (lane & 15);
  int k = kc * 32 + (e >> 3) * 16 + (lane >> 4) * 8 + (e & 7);
  dst[o] = bf1(W[(size_t)n * DIMD + k]);
}

// ---------------------------------------------------------------------------
// Deterministic mean over p: 56 chunks of 56 positions, then reduce.
// ---------------------------------------------------------------------------
__global__ void partial_mean(const float* __restrict__ x, const float* __restrict__ y,
                             float* __restrict__ px, float* __restrict__ py) {
  int b = blockIdx.x, chunk = blockIdx.y, d = threadIdx.x;
  float sx = 0.f, sy = 0.f;
  int p0 = chunk * 56;
  for (int i = 0; i < 56; ++i) {
    size_t idx = ((size_t)(b * NP + p0 + i)) * DIMD + d;
    sx += x[idx];
    sy += y[idx];
  }
  px[(b * 56 + chunk) * DIMD + d] = sx;
  py[(b * 56 + chunk) * DIMD + d] = sy;
}

__global__ void finalize_mean(const float* __restrict__ px, const float* __restrict__ py,
                              float* __restrict__ means) {
  int tid = blockIdx.x * blockDim.x + threadIdx.x;   // 0..4095 -> [2][8][256]
  int sel = tid >> 11;
  int rem = tid & 2047;
  int b = rem >> 8, d = rem & 255;
  const float* src = sel ? py : px;
  float s = 0.f;
  for (int c = 0; c < 56; ++c) s += src[(b * 56 + c) * DIMD + d];
  means[tid] = s * (1.0f / 3136.0f);
}

// ---------------------------------------------------------------------------
// Main fused kernel. 8 waves/WG; each wave owns 4 positions (16 M-rows).
// LDS per wave (40KB): z rows bf16 16x256 | qkv bf16 16x768 | attn-out bf16 16x256
// ---------------------------------------------------------------------------
__global__ __launch_bounds__(256, 1)
void attn_main(const float* __restrict__ x, const float* __restrict__ y,
               const float* __restrict__ means,
               const unsigned short* __restrict__ wq0, const unsigned short* __restrict__ wo0,
               const float* __restrict__ b0,
               const unsigned short* __restrict__ wq1, const unsigned short* __restrict__ wo1,
               const float* __restrict__ b1,
               float* __restrict__ out) {
  extern __shared__ char smem_raw[];
  const int wave = threadIdx.x >> 5;
  const int lane = threadIdx.x & 31;
  char* base = smem_raw + (size_t)wave * 40960;
  unsigned short* zbuf = (unsigned short*)base;                    // 16 x 256
  unsigned short* qkvb = (unsigned short*)(base + 8192);           // 16 x 768
  unsigned short* obuf = (unsigned short*)(base + 8192 + 24576);   // 16 x 256

  const int groupBase = (blockIdx.x * 8 + wave) * 4;  // first of 4 positions
  const int half = lane >> 4;
  const int mrow = lane & 15;

  // ---- build stage-1 input rows (f32 -> bf16 into LDS) ----
  for (int m = 0; m < 16; ++m) {
    int pos = m >> 2, ch = m & 3;
    int P = groupBase + pos;
    const float* src;
    if (ch == 0)      src = x + (size_t)P * DIMD;
    else if (ch == 1) src = y + (size_t)P * DIMD;
    else              src = means + ((size_t)(ch - 2) * NB + (P / NP)) * DIMD;
    for (int c = lane * 2; c < DIMD; c += 64) {
      *(unsigned*)&zbuf[m * DIMD + c] = pack_bf16(src[c], src[c + 1]);
    }
  }
  __syncthreads();

  for (int stage = 0; stage < 2; ++stage) {
    const unsigned short* Wq = stage ? wq1 : wq0;
    const unsigned short* Wo = stage ? wo1 : wo0;
    const float*         bia = stage ? b1  : b0;

    // ---- A fragments from zbuf (16x256 bf16, 8 K-chunks) ----
    AF a[8];
    #pragma unroll
    for (int kc = 0; kc < 8; ++kc) {
      #pragma unroll
      for (int j = 0; j < 8; ++j) {
        int k = kc * 32 + ((j >= 4) ? 16 : 0) + half * 8 + (j & 3) * 2;
        a[kc].u[j] = *(const unsigned*)&zbuf[mrow * DIMD + k];
      }
    }

    // ---- qkv = z @ Wqkv^T : 48 N-tiles, two tiles per step (2 indep chains) --
    #pragma unroll 1
    for (int tt = 0; tt < 24; ++tt) {
      int t0 = tt * 2, t1 = tt * 2 + 1;
      v8f c0 = {}, c1 = {};
      #pragma unroll
      for (int kc = 0; kc < 8; ++kc) {
        const v16bf bf0 = *(const v16bf*)(Wq + (((size_t)t0 * 8 + kc) * 32 + lane) * 16);
        const v16bf bf1v = *(const v16bf*)(Wq + (((size_t)t1 * 8 + kc) * 32 + lane) * 16);
        c0 = __builtin_amdgcn_wmma_f32_16x16x32_bf16(false, a[kc].v, false, bf0,
                                                     (short)0, c0, false, false);
        c1 = __builtin_amdgcn_wmma_f32_16x16x32_bf16(false, a[kc].v, false, bf1v,
                                                     (short)0, c1, false, false);
      }
      int n0 = t0 * 16 + mrow, n1 = t1 * 16 + mrow;
      #pragma unroll
      for (int r = 0; r < 8; ++r) {
        int m = half * 8 + r;
        qkvb[m * QKVN + n0] = bf1(c0[r]);
        qkvb[m * QKVN + n1] = bf1(c1[r]);
      }
    }
    __syncthreads();

    // ---- tiny attention: one (position, head) per lane ----
    {
      const int apos = lane >> 3;   // 0..3
      const int head = lane & 7;    // 0..7
      unsigned kreg[4][16];
      #pragma unroll
      for (int j = 0; j < 4; ++j)
        #pragma unroll
        for (int d2 = 0; d2 < 16; ++d2)
          kreg[j][d2] = *(const unsigned*)&qkvb[(apos * 4 + j) * QKVN + 256 + head * 32 + d2 * 2];

      float prob[4][4];
      #pragma unroll
      for (int i = 0; i < 4; ++i) {
        float s[4] = {0.f, 0.f, 0.f, 0.f};
        #pragma unroll
        for (int d2 = 0; d2 < 16; ++d2) {
          unsigned q = *(const unsigned*)&qkvb[(apos * 4 + i) * QKVN + head * 32 + d2 * 2];
          float ql = bflo(q), qh = bfhi(q);
          #pragma unroll
          for (int j = 0; j < 4; ++j) {
            s[j] = fmaf(ql, bflo(kreg[j][d2]), s[j]);
            s[j] = fmaf(qh, bfhi(kreg[j][d2]), s[j]);
          }
        }
        #pragma unroll
        for (int j = 0; j < 4; ++j) {
          float mk;
          if (stage == 0) mk = (j == 3 - i) ? -100.f : 0.f;                       // MASK0
          else            mk = ((i >= 2) || (((i + j) & 1) != 0)) ? -100.f : 0.f; // MASK1
          s[j] = s[j] * SCALE_F + mk;
        }
        float mx = fmaxf(fmaxf(s[0], s[1]), fmaxf(s[2], s[3]));
        float e0 = __expf(s[0] - mx), e1 = __expf(s[1] - mx);
        float e2 = __expf(s[2] - mx), e3 = __expf(s[3] - mx);
        float rs = 1.f / (e0 + e1 + e2 + e3);
        prob[i][0] = e0 * rs; prob[i][1] = e1 * rs;
        prob[i][2] = e2 * rs; prob[i][3] = e3 * rs;
      }

      // out = attn @ v  -> obuf rows (bf16)
      #pragma unroll
      for (int d2 = 0; d2 < 16; ++d2) {
        unsigned vv[4];
        #pragma unroll
        for (int j = 0; j < 4; ++j)
          vv[j] = *(const unsigned*)&qkvb[(apos * 4 + j) * QKVN + 512 + head * 32 + d2 * 2];
        #pragma unroll
        for (int i = 0; i < 4; ++i) {
          float lo = 0.f, hi = 0.f;
          #pragma unroll
          for (int j = 0; j < 4; ++j) {
            lo = fmaf(prob[i][j], bflo(vv[j]), lo);
            hi = fmaf(prob[i][j], bfhi(vv[j]), hi);
          }
          *(unsigned*)&obuf[(apos * 4 + i) * DIMD + head * 32 + d2 * 2] = pack_bf16(lo, hi);
        }
      }
    }
    __syncthreads();

    // ---- out-proj: A from obuf, 16 N-tiles, two per step ----
    AF ao[8];
    #pragma unroll
    for (int kc = 0; kc < 8; ++kc) {
      #pragma unroll
      for (int j = 0; j < 8; ++j) {
        int k = kc * 32 + ((j >= 4) ? 16 : 0) + half * 8 + (j & 3) * 2;
        ao[kc].u[j] = *(const unsigned*)&obuf[mrow * DIMD + k];
      }
    }

    #pragma unroll 1
    for (int tt = 0; tt < 8; ++tt) {
      int t0 = tt * 2, t1 = tt * 2 + 1;
      v8f c0 = {}, c1 = {};
      #pragma unroll
      for (int kc = 0; kc < 8; ++kc) {
        const v16bf bf0 = *(const v16bf*)(Wo + (((size_t)t0 * 8 + kc) * 32 + lane) * 16);
        const v16bf bf1v = *(const v16bf*)(Wo + (((size_t)t1 * 8 + kc) * 32 + lane) * 16);
        c0 = __builtin_amdgcn_wmma_f32_16x16x32_bf16(false, ao[kc].v, false, bf0,
                                                     (short)0, c0, false, false);
        c1 = __builtin_amdgcn_wmma_f32_16x16x32_bf16(false, ao[kc].v, false, bf1v,
                                                     (short)0, c1, false, false);
      }
      int n0 = t0 * 16 + mrow, n1 = t1 * 16 + mrow;
      float bv0 = bia[n0], bv1 = bia[n1];
      if (stage == 0) {
        #pragma unroll
        for (int r = 0; r < 8; ++r) {
          int m = half * 8 + r;
          zbuf[m * DIMD + n0] = bf1(c0[r] + bv0);   // stage-2 input
          zbuf[m * DIMD + n1] = bf1(c1[r] + bv1);
        }
      } else {
        #pragma unroll
        for (int r = 0; r < 8; ++r) {
          int m = half * 8 + r;
          int ch = m & 3;
          if (ch < 2) {                             // x1 / y1 only
            int P = groupBase + (m >> 2);
            out[(size_t)ch * OUTPLANE + (size_t)P * DIMD + n0] = c0[r] + bv0;
            out[(size_t)ch * OUTPLANE + (size_t)P * DIMD + n1] = c1[r] + bv1;
          }
        }
      }
    }
    __syncthreads();
  }
}

// ---------------------------------------------------------------------------
// Host-side launch. Workspace layout (bytes):
//   [0,        393216)  wq0 bf16   (768x256 packed B-fragments)
//   [393216,   524288)  wo0 bf16   (256x256)
//   [524288,   917504)  wq1 bf16
//   [917504,  1048576)  wo1 bf16
//   [1048576, 1064960)  means f32  [2][8][256]
//   [1064960, 1523712)  partial x  [8][56][256] f32
//   [1523712, 1982464)  partial y  [8][56][256] f32
// ---------------------------------------------------------------------------
extern "C" void kernel_launch(void* const* d_in, const int* in_sizes, int n_in,
                              void* d_out, int out_size, void* d_ws, size_t ws_size,
                              hipStream_t stream) {
  const float* x     = (const float*)d_in[0];
  const float* y     = (const float*)d_in[1];
  const float* Wqkv0 = (const float*)d_in[2];
  const float* Wout0 = (const float*)d_in[3];
  const float* bout0 = (const float*)d_in[4];
  const float* Wqkv1 = (const float*)d_in[5];
  const float* Wout1 = (const float*)d_in[6];
  const float* bout1 = (const float*)d_in[7];
  float* out = (float*)d_out;

  char* ws = (char*)d_ws;
  unsigned short* wq0 = (unsigned short*)(ws + 0);
  unsigned short* wo0 = (unsigned short*)(ws + 393216);
  unsigned short* wq1 = (unsigned short*)(ws + 524288);
  unsigned short* wo1 = (unsigned short*)(ws + 917504);
  float* means = (float*)(ws + 1048576);
  float* px    = (float*)(ws + 1064960);
  float* py    = (float*)(ws + 1523712);

  pack_weights<<<768, 256, 0, stream>>>(Wqkv0, wq0, 48);
  pack_weights<<<256, 256, 0, stream>>>(Wout0, wo0, 16);
  pack_weights<<<768, 256, 0, stream>>>(Wqkv1, wq1, 48);
  pack_weights<<<256, 256, 0, stream>>>(Wout1, wo1, 16);

  partial_mean<<<dim3(8, 56), 256, 0, stream>>>(x, y, px, py);
  finalize_mean<<<16, 256, 0, stream>>>(px, py, means);

  // 784 WGs x 8 waves x 4 positions = 25088 positions; 320KB dynamic LDS/WG.
  attn_main<<<784, 256, 327680, stream>>>(x, y, means, wq0, wo0, bout0,
                                          wq1, wo1, bout1, out);
}
